// Net_49641232007795
// MI455X (gfx1250) — matrix operand order repaired and checked by reading, not compiled, not run on hardware
//
#include <hip/hip_runtime.h>

// ---------------------------------------------------------------------------
// Types
// ---------------------------------------------------------------------------
typedef _Float16 v16h __attribute__((ext_vector_type(16)));
typedef _Float16 v8h  __attribute__((ext_vector_type(8)));
typedef float    v8f  __attribute__((ext_vector_type(8)));
typedef float    v4f  __attribute__((ext_vector_type(4)));

#define DI __device__ __forceinline__

// ---------------------------------------------------------------------------
// Fast tanh: gfx1250 has V_TANH_F32 (TRANS unit, co-executes with WMMA/XDL)
// ---------------------------------------------------------------------------
DI float fast_tanh(float x) {
#if __has_builtin(__builtin_amdgcn_tanhf)
  return __builtin_amdgcn_tanhf(x);
#elif __has_builtin(__builtin_amdgcn_tanh_f32)
  return __builtin_amdgcn_tanh_f32(x);
#else
  float ax = fminf(fabsf(x), 9.0f);
  float e  = __expf(2.0f * ax);
  float t  = 1.0f - 2.0f / (e + 1.0f);
  return copysignf(t, x);
#endif
}

// ---------------------------------------------------------------------------
// WMMA f32 = f16 x f16 + f32   (16x16x32)
// ---------------------------------------------------------------------------
DI v8f wmma(v16h a, v16h b, v8f c) {
  return __builtin_amdgcn_wmma_f32_16x16x32_f16(false, a, false, b, (short)0, c,
                                                false, false);
}

// A-fragment (16x32, f16 source, row-major, lane l: row=l&15, g=l>>4):
//   e0..7  -> K = kb + g*8 + e          (16B contiguous)
//   e8..15 -> K = kb + 16 + g*8 + (e-8) (16B contiguous)
DI v16h ldA_h(const _Float16* row, int kb, int g) {
  v8h lo = *(const v8h*)(row + kb + g * 8);
  v8h hi = *(const v8h*)(row + kb + 16 + g * 8);
  return __builtin_shufflevector(lo, hi, 0, 1, 2, 3, 4, 5, 6, 7, 8, 9, 10, 11,
                                 12, 13, 14, 15);
}

DI v8h cvt8f(const float* p) {
  v4f a = *(const v4f*)p;
  v4f b = *(const v4f*)(p + 4);
  v8h r;
#pragma unroll
  for (int i = 0; i < 4; ++i) {
    r[i]     = (_Float16)a[i];
    r[i + 4] = (_Float16)b[i];
  }
  return r;
}

// A-fragment loaded straight from fp32 global memory (converted in-register)
DI v16h ldA_f32(const float* row, int kb, int g) {
  v8h lo = cvt8f(row + kb + g * 8);
  v8h hi = cvt8f(row + kb + 16 + g * 8);
  return __builtin_shufflevector(lo, hi, 0, 1, 2, 3, 4, 5, 6, 7, 8, 9, 10, 11,
                                 12, 13, 14, 15);
}

// B-fragment (32x16): lane l holds column (l&15), e0..7 -> K = kb + g*8 + e,
// e8..15 -> K = kb + 16 + g*8 + (e-8) (two 16B loads per lane).
DI v16h ldB(const _Float16* wrow, int kb, int g) {
  v8h lo = *(const v8h*)(wrow + kb + g * 8);
  v8h hi = *(const v8h*)(wrow + kb + 16 + g * 8);
  return __builtin_shufflevector(lo, hi, 0, 1, 2, 3, 4, 5, 6, 7, 8, 9, 10, 11,
                                 12, 13, 14, 15);
}

// ---------------------------------------------------------------------------
// fp32 -> f16 conversion (weights)
// ---------------------------------------------------------------------------
__global__ void cvt_kernel(const float* __restrict__ src,
                           _Float16* __restrict__ dst, size_t n) {
  size_t i  = (size_t)blockIdx.x * blockDim.x + threadIdx.x;
  size_t st = (size_t)gridDim.x * blockDim.x;
  for (; i < n; i += st) dst[i] = (_Float16)src[i];
}

// ---------------------------------------------------------------------------
// Tree kernel.
// Dead-code analysis of the reference: block_out = prev[:,:,35] depends only on
// nodes {leaf3..7, 11,12,13,14, 19,20, 21, 27,28, 35} -> 15 nodes, remapped to
// slots 0..14. Every node is: h = tanh(A(16x64) @ W1^T(64x128)),
//                             o = tanh(h @ W2^T(128x32)).
// One wave / workgroup, M = 32 (two row tiles share all B fragments).
// ---------------------------------------------------------------------------
#define PSTR 40   // prev buffer row stride (halves), bank-conflict-free
#define HSTR 136  // hidden buffer row stride (halves), bank-conflict-free

__device__ const int T_c[10]  = {0, 0, 0, 0, 7, 7, 13, 18, 18, 27};  // level cum
__device__ const int T_m[10]  = {7, 7, 7, 7, 6, 6, 5, 4, 4, 1};      // 8-d
__device__ const int T_j[10]  = {3, 4, 5, 6, 4, 5, 0, 1, 2, 0};      // node j
__device__ const int T_ja[10] = {0, 1, 2, 3, 5, 6, 7, 9, 10, 12};    // src slot

__global__ __launch_bounds__(32) void tree_kernel(
    const float* __restrict__ x, const _Float16* __restrict__ lw1,
    const _Float16* __restrict__ lw2, const _Float16* __restrict__ iw1,
    const _Float16* __restrict__ iw2, _Float16* __restrict__ bout) {
  __shared__ _Float16 s_prev[2 * 15 * 16 * PSTR];  // 38400 B
  __shared__ _Float16 s_hid[2 * 16 * HSTR];        //  8704 B

  const int l  = threadIdx.x;
  const int lr = l & 15;
  const int g  = l >> 4;
  const int n  = blockIdx.x & 63;   // block
  const int bt = blockIdx.x >> 6;   // batch tile of 32 rows
  const int row0 = bt * 32;

#pragma unroll 1
  for (int slot = 0; slot < 15; ++slot) {
    v16h a[2][2];
    const _Float16 *w1, *w2;
    if (slot < 5) {
      const int p       = 3 + slot;                       // live leaves 3..7
      const float* xb   = x + (size_t)(n * 8 + p) * 64;
#pragma unroll
      for (int rt = 0; rt < 2; ++rt) {
        const float* xr = xb + (size_t)(row0 + rt * 16 + lr) * 32768;
        a[rt][0] = ldA_f32(xr, 0, g);
        a[rt][1] = ldA_f32(xr, 32, g);
      }
      w1 = lw1 + (size_t)(n * 8 + p) * 8192;
      w2 = lw2 + (size_t)(n * 8 + p) * 4096;
    } else {
      const int t  = slot - 5;
      const int ja = T_ja[t];
#pragma unroll
      for (int rt = 0; rt < 2; ++rt) {
        const _Float16* pa = s_prev + ((rt * 15 + ja) * 16 + lr) * PSTR;
        const _Float16* pb = s_prev + ((rt * 15 + ja + 1) * 16 + lr) * PSTR;
        a[rt][0] = ldA_h(pa, 0, g);
        a[rt][1] = ldA_h(pb, 0, g);
      }
      const size_t nd = (size_t)(n * T_m[t] + T_j[t]);
      w1 = iw1 + (size_t)T_c[t] * 524288 + nd * 8192;
      w2 = iw2 + (size_t)T_c[t] * 262144 + nd * 4096;
    }

    // GEMM1: hidden(16x128) = tanh(A(16x64) @ W1^T), W1 rows = 128 outs, K=64
#pragma unroll
    for (int ct = 0; ct < 8; ++ct) {
      const _Float16* wr = w1 + (ct * 16 + lr) * 64;
      v16h b0 = ldB(wr, 0, g);
      v16h b1 = ldB(wr, 32, g);
#pragma unroll
      for (int rt = 0; rt < 2; ++rt) {
        v8f acc = {};
        acc = wmma(a[rt][0], b0, acc);
        acc = wmma(a[rt][1], b1, acc);
#pragma unroll
        for (int i = 0; i < 8; ++i)
          s_hid[(rt * 16 + g * 8 + i) * HSTR + ct * 16 + lr] =
              (_Float16)fast_tanh(acc[i]);
      }
    }

    // GEMM2: out(16x32) = tanh(hidden @ W2^T), W2 rows = 32 outs, K=128
    v16h ha[2][4];
#pragma unroll
    for (int rt = 0; rt < 2; ++rt)
#pragma unroll
      for (int kt = 0; kt < 4; ++kt)
        ha[rt][kt] = ldA_h(s_hid + (rt * 16 + lr) * HSTR, kt * 32, g);

#pragma unroll
    for (int c2 = 0; c2 < 2; ++c2) {
      v16h b[4];
#pragma unroll
      for (int kt = 0; kt < 4; ++kt)
        b[kt] = ldB(w2 + (c2 * 16 + lr) * 128, kt * 32, g);
#pragma unroll
      for (int rt = 0; rt < 2; ++rt) {
        v8f acc = {};
#pragma unroll
        for (int kt = 0; kt < 4; ++kt) acc = wmma(ha[rt][kt], b[kt], acc);
#pragma unroll
        for (int i = 0; i < 8; ++i)
          s_prev[((rt * 15 + slot) * 16 + g * 8 + i) * PSTR + c2 * 16 + lr] =
              (_Float16)fast_tanh(acc[i]);
      }
    }
  }

  // Write block_out rows from slot 14 (= node 35). Lane -> (rowtile g, row lr)
  {
    const _Float16* src = s_prev + ((g * 15 + 14) * 16 + lr) * PSTR;
    _Float16* dst = bout + (size_t)(row0 + g * 16 + lr) * 2048 + n * 32;
#pragma unroll
    for (int q = 0; q < 4; ++q) ((v8h*)dst)[q] = ((const v8h*)src)[q];
  }
}

// ---------------------------------------------------------------------------
// Cat GEMM: rc(2048x1024) = tanh(block_out(2048x2048) @ cat_w^T)
// Wave tile 32x64, K = 2048.
// ---------------------------------------------------------------------------
__global__ __launch_bounds__(32) void cat_kernel(
    const _Float16* __restrict__ A, const _Float16* __restrict__ W,
    _Float16* __restrict__ rc) {
  const int l  = threadIdx.x;
  const int lr = l & 15;
  const int g  = l >> 4;
  const int r0 = (blockIdx.x & 63) * 32;
  const int c0 = (blockIdx.x >> 6) * 64;

  v8f acc[2][4] = {};
#pragma unroll 2
  for (int kt = 0; kt < 64; ++kt) {
    const int kb = kt * 32;
    v16h a0 = ldA_h(A + (size_t)(r0 + lr) * 2048, kb, g);
    v16h a1 = ldA_h(A + (size_t)(r0 + 16 + lr) * 2048, kb, g);
#pragma unroll
    for (int cc = 0; cc < 4; ++cc) {
      v16h b = ldB(W + (size_t)(c0 + cc * 16 + lr) * 2048, kb, g);
      acc[0][cc] = wmma(a0, b, acc[0][cc]);
      acc[1][cc] = wmma(a1, b, acc[1][cc]);
    }
  }
#pragma unroll
  for (int rt = 0; rt < 2; ++rt)
#pragma unroll
    for (int cc = 0; cc < 4; ++cc)
#pragma unroll
      for (int i = 0; i < 8; ++i)
        rc[(size_t)(r0 + rt * 16 + g * 8 + i) * 1024 + c0 + cc * 16 + lr] =
            (_Float16)fast_tanh(acc[rt][cc][i]);
}

// ---------------------------------------------------------------------------
// Head: a = softmax(relu(rc @ act_w^T)), v = relu(rc @ val_w^T)
// One wave handles 16 rows.
// ---------------------------------------------------------------------------
__global__ __launch_bounds__(32) void head_kernel(
    const _Float16* __restrict__ rc, const _Float16* __restrict__ aw,
    const _Float16* __restrict__ vw, float* __restrict__ outa,
    float* __restrict__ outv) {
  __shared__ float sc[16 * 32];
  __shared__ float part[32];
  const int l  = threadIdx.x;
  const int lr = l & 15;
  const int g  = l >> 4;
  const int r0 = blockIdx.x * 16;

  v8f s0 = {}, s1 = {};
#pragma unroll 4
  for (int kt = 0; kt < 32; ++kt) {
    const int kb = kt * 32;
    v16h a  = ldA_h(rc + (size_t)(r0 + lr) * 1024, kb, g);
    v16h b0 = ldB(aw + lr * 1024, kb, g);
    v16h b1 = ldB(aw + (16 + lr) * 1024, kb, g);
    s0 = wmma(a, b0, s0);
    s1 = wmma(a, b1, s1);
  }
#pragma unroll
  for (int i = 0; i < 8; ++i) {
    const int row = g * 8 + i;
    sc[row * 32 + lr]      = fmaxf(s0[i], 0.0f);
    sc[row * 32 + 16 + lr] = fmaxf(s1[i], 0.0f);
  }

  // value head: 2 lanes per row, 512-long partial dots each
  float vp = 0.0f;
  const _Float16* rp = rc + (size_t)(r0 + lr) * 1024 + g * 512;
  const _Float16* wp = vw + g * 512;
#pragma unroll 4
  for (int q = 0; q < 64; ++q) {
    v8h rv = *(const v8h*)(rp + q * 8);
    v8h vv = *(const v8h*)(wp + q * 8);
#pragma unroll
    for (int i = 0; i < 8; ++i) vp += (float)rv[i] * (float)vv[i];
  }
  part[l] = vp;
  __syncthreads();

  if (l < 16) {
    float m = 0.0f;  // scores are >= 0 after relu
#pragma unroll
    for (int c = 0; c < 32; ++c) m = fmaxf(m, sc[l * 32 + c]);
    float s = 0.0f;
#pragma unroll
    for (int c = 0; c < 32; ++c) {
      float e = __expf(sc[l * 32 + c] - m);
      sc[l * 32 + c] = e;
      s += e;
    }
    const float inv = 1.0f / s;
#pragma unroll
    for (int c = 0; c < 32; ++c)
      outa[(size_t)(r0 + l) * 32 + c] = sc[l * 32 + c] * inv;
    outv[r0 + l] = fmaxf(part[l] + part[l + 16], 0.0f);
  }
}

// ---------------------------------------------------------------------------
// Launch
// ---------------------------------------------------------------------------
extern "C" void kernel_launch(void* const* d_in, const int* in_sizes, int n_in,
                              void* d_out, int out_size, void* d_ws,
                              size_t ws_size, hipStream_t stream) {
  (void)in_sizes;
  (void)out_size;
  (void)ws_size;

  constexpr size_t N_LW1 = 64ull * 8 * 128 * 64;   // 4194304
  constexpr size_t N_LW2 = 64ull * 8 * 32 * 128;   // 2097152
  constexpr size_t N_IW1 = 64ull * 28 * 128 * 64;  // 14680064
  constexpr size_t N_IW2 = 64ull * 28 * 32 * 128;  // 7340032
  constexpr size_t N_CAT = 1024ull * 2048;
  constexpr size_t N_ACT = 32ull * 1024;
  constexpr size_t N_VAL = 1024ull;

  _Float16* ws    = (_Float16*)d_ws;
  _Float16* h_lw1 = ws;
  _Float16* h_lw2 = h_lw1 + N_LW1;
  _Float16* h_iw1 = h_lw2 + N_LW2;
  _Float16* h_iw2 = h_iw1 + N_IW1;
  _Float16* h_cat = h_iw2 + N_IW2;
  _Float16* h_act = h_cat + N_CAT;
  _Float16* h_val = h_act + N_ACT;
  _Float16* h_bo  = h_val + N_VAL;          // block_out 2048x2048
  _Float16* h_rc  = h_bo + 2048ull * 2048;  // rc 2048x1024

  const float* x = (const float*)d_in[0];

  auto cvt = [&](const void* s, _Float16* d, size_t n) {
    cvt_kernel<<<1024, 256, 0, stream>>>((const float*)s, d, n);
  };

  cvt(d_in[1], h_lw1, N_LW1);
  cvt(d_in[2], h_lw2, N_LW2);
  if (n_in >= 20) {
    // tuples flattened: inner_w1 -> d_in[3..9], inner_w2 -> d_in[10..16]
    size_t o1 = 0, o2 = 0;
    for (int d = 1; d <= 7; ++d) {
      const size_t n1 = 64ull * (8 - d) * 128 * 64;
      const size_t n2 = 64ull * (8 - d) * 32 * 128;
      cvt(d_in[2 + d], h_iw1 + o1, n1);
      cvt(d_in[9 + d], h_iw2 + o2, n2);
      o1 += n1;
      o2 += n2;
    }
    cvt(d_in[17], h_cat, N_CAT);
    cvt(d_in[18], h_act, N_ACT);
    cvt(d_in[19], h_val, N_VAL);
  } else {
    // tuples passed pre-concatenated
    cvt(d_in[3], h_iw1, N_IW1);
    cvt(d_in[4], h_iw2, N_IW2);
    cvt(d_in[5], h_cat, N_CAT);
    cvt(d_in[6], h_act, N_ACT);
    cvt(d_in[7], h_val, N_VAL);
  }

  // Tree: 64 batch tiles (32 rows) x 64 blocks, one wave each.
  tree_kernel<<<dim3(4096), dim3(32), 0, stream>>>(x, h_lw1, h_lw2, h_iw1,
                                                   h_iw2, h_bo);
  // Cat GEMM: 64 row tiles x 16 col tiles.
  cat_kernel<<<dim3(1024), dim3(32), 0, stream>>>(h_bo, h_cat, h_rc);
  // Head: 128 x 16-row tiles.
  head_kernel<<<dim3(128), dim3(32), 0, stream>>>(
      h_rc, h_act, h_val, (float*)d_out, (float*)d_out + 2048ull * 32);
}
